// SoftGroupAttention_56822417326256
// MI455X (gfx1250) — matrix operand order repaired and checked by reading, not compiled, not run on hardware
//
#include <hip/hip_runtime.h>

// ---------------------------------------------------------------------------
// SoftGroupAttention fused CDNA5 implementation (bf16 WMMA, f32 accumulate)
// B=16, N=1024, DIM=384, GP=49 (padded->64), TOPK = int(1024*0.2) = 204
// All hot-loop operands are contiguous 32B bf16 loads; fp32->bf16 conversion
// and weight transposition happen once in prep kernels. GEMM inner loops use
// explicit one-ahead fragment rotation so the next b128 loads stay live
// across the current WMMA (double-buffered instead of load->wait0->wmma).
// ---------------------------------------------------------------------------

#define B_    16
#define N_    1024
#define DIM_  384
#define QKVG_ 1536
#define GPP_  64      // padded group count
#define TOPK_ 204

typedef __attribute__((ext_vector_type(16))) __bf16 v16bf;
typedef __attribute__((ext_vector_type(8)))  float  v8f;

union V16 { v16bf v; uint4 u[2]; unsigned short s[16]; };
union U4w { uint4 u; unsigned short s[8]; };

__device__ __forceinline__ unsigned short f2bf(float f) {
    unsigned int u = __float_as_uint(f);
    unsigned int r = u + 0x7FFFu + ((u >> 16) & 1u);   // round-to-nearest-even
    return (unsigned short)(r >> 16);
}

__device__ __forceinline__ v8f zero8() {
    v8f c;
#pragma unroll
    for (int i = 0; i < 8; ++i) c[i] = 0.0f;
    return c;
}

// A fragment (16x32 bf16, MxK): lane holds row m=lane&15; K-halves
// [kk+abase .. +15], abase = 16 for upper half-wave.
__device__ __forceinline__ v16bf load_a_bf(const unsigned short* base, int ldRow,
                                           int kk, int lane) {
    const unsigned short* row = base + (size_t)(lane & 15) * ldRow + kk + ((lane & 16) ? 16 : 0);
    V16 r;
    r.u[0] = *(const uint4*)(row);
    r.u[1] = *(const uint4*)(row + 8);
    return r.v;
}

// B fragment (32x16 bf16, KxN) from K-contiguous storage: column n = lane&15
// is a row of the source; 16 contiguous K-halves per lane.
__device__ __forceinline__ v16bf load_b_kcontig(const unsigned short* base, int ldRow,
                                                int kk, int lane) {
    const unsigned short* row = base + (size_t)(lane & 15) * ldRow + kk + ((lane & 16) ? 16 : 0);
    V16 r;
    r.u[0] = *(const uint4*)(row);
    r.u[1] = *(const uint4*)(row + 8);
    return r.v;
}

#define WMMA_BF16(c, a, b) \
    __builtin_amdgcn_wmma_f32_16x16x32_bf16(false, (a), false, (b), (short)0, (c), false, false)

// ---------------------------------------------------------------------------
// Prep: x (f32) -> bf16, 4 elements/thread, packed stores
// ---------------------------------------------------------------------------
__global__ __launch_bounds__(256) void k_prep_x(const float* __restrict__ x,
                                                unsigned short* __restrict__ xbf) {
    int i = blockIdx.x * 256 + threadIdx.x;            // total/4 threads exactly
    float4 f = ((const float4*)x)[i];
    uint2 o;
    o.x = (unsigned int)f2bf(f.x) | ((unsigned int)f2bf(f.y) << 16);
    o.y = (unsigned int)f2bf(f.z) | ((unsigned int)f2bf(f.w) << 16);
    ((uint2*)xbf)[i] = o;
}

// ---------------------------------------------------------------------------
// Prep: W [K x Ncols] f32 -> WT [NcolsPad x K] bf16 (transposed, zero-padded)
// ---------------------------------------------------------------------------
__global__ __launch_bounds__(128) void k_prep_wT(const float* __restrict__ W,
                                                 unsigned short* __restrict__ WT,
                                                 int K, int Ncols) {
    int n = blockIdx.x;                                 // output row (= input col)
    for (int k = threadIdx.x; k < K; k += 128) {
        float f = (n < Ncols) ? W[(size_t)k * Ncols + n] : 0.0f;
        WT[(size_t)n * K + k] = f2bf(f);
    }
}

// ---------------------------------------------------------------------------
// Pipelined 16x(8*16) x K=384 GEMM body shared by k_qkvg / k_proj:
// linearized (j,t) loop with one-ahead B rotation + A rotation per j.
// ---------------------------------------------------------------------------
__device__ __forceinline__ void gemm_8tile_pipelined(const unsigned short* arows,
                                                     const unsigned short* wbase,
                                                     int lane, v8f acc[8]) {
    v16bf a    = load_a_bf(arows, DIM_, 0, lane);
    v16bf bcur = load_b_kcontig(wbase, DIM_, 0, lane);
#pragma unroll
    for (int j = 0; j < 12; ++j) {
#pragma unroll
        for (int t = 0; t < 8; ++t) {
            int nidx = j * 8 + t + 1; if (nidx > 95) nidx = 95;   // tail: cached repeat
            int jn = nidx >> 3, tn = nidx & 7;
            v16bf bnext = load_b_kcontig(wbase + (size_t)(tn * 16) * DIM_, DIM_, jn * 32, lane);
            v16bf anext = (t == 7) ? load_a_bf(arows, DIM_, (j < 11 ? j + 1 : 11) * 32, lane) : a;
            acc[t] = WMMA_BF16(acc[t], a, bcur);
            bcur = bnext;
            a = anext;
        }
    }
}

// ---------------------------------------------------------------------------
// Kernel 1: qkvg = x @ W_qkvg + b  -> bf16 [B*N, 1536]
// Block = 1 wave computing 8 column tiles (128 cols); grid (12, 1024).
// ---------------------------------------------------------------------------
__global__ __launch_bounds__(32) void k_qkvg(const unsigned short* __restrict__ xbf,
                                             const unsigned short* __restrict__ WT,
                                             const float* __restrict__ bias,
                                             unsigned short* __restrict__ qkvg) {
    int lane = threadIdx.x;
    int cb = blockIdx.x, rt = blockIdx.y;
    const unsigned short* xrows = xbf + (size_t)rt * 16 * DIM_;
    const unsigned short* wbase = WT + (size_t)(cb * 128) * DIM_;

    v8f acc[8];
#pragma unroll
    for (int t = 0; t < 8; ++t) {
        float bv = bias[cb * 128 + t * 16 + (lane & 15)];
#pragma unroll
        for (int i = 0; i < 8; ++i) acc[t][i] = bv;
    }
    gemm_8tile_pipelined(xrows, wbase, lane, acc);

    unsigned short* orows = qkvg + (size_t)rt * 16 * QKVG_;
    int m0 = (lane & 16) ? 8 : 0, n = lane & 15;
#pragma unroll
    for (int t = 0; t < 8; ++t)
#pragma unroll
        for (int r = 0; r < 8; ++r)
            orows[(size_t)(r + m0) * QKVG_ + cb * 128 + t * 16 + n] = f2bf(acc[t][r]);
}

// ---------------------------------------------------------------------------
// Kernel 2: gw = softmax(g @ W_gp) -> bf16 [B*N, 64] (cols 49..63 zero)
// grid 1024 row tiles, block = 1 wave (4 column tiles + in-wave softmax)
// ---------------------------------------------------------------------------
__global__ __launch_bounds__(32) void k_gw(const unsigned short* __restrict__ qkvg,
                                           const unsigned short* __restrict__ WgpT,
                                           unsigned short* __restrict__ gwout) {
    int lane = threadIdx.x;
    int rt = blockIdx.x;
    const unsigned short* grows = qkvg + (size_t)rt * 16 * QKVG_ + 3 * DIM_;

    v8f c[4];
#pragma unroll
    for (int t = 0; t < 4; ++t) c[t] = zero8();
#pragma unroll
    for (int j = 0; j < 12; ++j) {
        v16bf a = load_a_bf(grows, QKVG_, j * 32, lane);
        v16bf b[4];
#pragma unroll
        for (int t = 0; t < 4; ++t)
            b[t] = load_b_kcontig(WgpT + (size_t)(t * 16) * DIM_, DIM_, j * 32, lane);
#pragma unroll
        for (int t = 0; t < 4; ++t)
            c[t] = WMMA_BF16(c[t], a, b[t]);
    }

    int n = lane & 15, m0 = (lane & 16) ? 8 : 0;
#pragma unroll
    for (int r = 0; r < 8; ++r) {
        float mx = -1e30f;
#pragma unroll
        for (int t = 0; t < 4; ++t) {
            int col = t * 16 + n;
            mx = fmaxf(mx, (col < 49) ? c[t][r] : -1e30f);
        }
#pragma unroll
        for (int off = 1; off < 16; off <<= 1) mx = fmaxf(mx, __shfl_xor(mx, off, 32));
        float e[4], sum = 0.0f;
#pragma unroll
        for (int t = 0; t < 4; ++t) {
            int col = t * 16 + n;
            e[t] = (col < 49) ? __expf(c[t][r] - mx) : 0.0f;
            sum += e[t];
        }
#pragma unroll
        for (int off = 1; off < 16; off <<= 1) sum += __shfl_xor(sum, off, 32);
        float inv = 1.0f / sum;
        size_t row = (size_t)rt * 16 + r + m0;
#pragma unroll
        for (int t = 0; t < 4; ++t)
            gwout[row * GPP_ + t * 16 + n] = f2bf(e[t] * inv);
    }
}

// ---------------------------------------------------------------------------
// Kernel 3: fused attention per 16-row tile.
//  A: S = (Q Kᵀ)*scale -> LDS f32[16][1024]  (Q frags hoisted; dual-acc chain
//     with one-pair-ahead B rotation)
//  B: Gsim = GW GWᵀ    -> LDS f32[16][1024]
//  C: per-row softmax, exact top-204 threshold (binary search on nonneg float
//     bits, wave-uniform), mask + renormalize -> bf16 row (overlays Gsim LDS)
//  D: out = W @ V; V chunks staged transposed in LDS, shared by all 8 waves
// grid 1024 tiles, block 256 (8 waves), dynamic LDS 128 KB
// ---------------------------------------------------------------------------
__global__ __launch_bounds__(256) void k_attn(const unsigned short* __restrict__ qkvg,
                                              const unsigned short* __restrict__ gw,
                                              unsigned short* __restrict__ attn) {
    extern __shared__ char smem[];
    float* sS = (float*)smem;                      // 16*1024 f32 = 64KB (later: sV)
    float* sG = sS + 16 * N_;                      // 16*1024 f32 = 64KB (later: sW bf16)
    int lane = threadIdx.x & 31;
    int wave = threadIdx.x >> 5;
    int rt = blockIdx.x;
    int b  = rt >> 6;

    const unsigned short* qtile = qkvg + (size_t)rt * 16 * QKVG_;
    const unsigned short* kbase = qkvg + (size_t)b * N_ * QKVG_ + DIM_;
    const unsigned short* vbase = qkvg + (size_t)b * N_ * QKVG_ + 2 * DIM_;
    const unsigned short* gtile = gw + (size_t)rt * 16 * GPP_;
    const unsigned short* gall  = gw + (size_t)b * N_ * GPP_;
    const float scale = 0.051031036307982884f;     // 384^-0.5

    int m0 = (lane & 16) ? 8 : 0, n = lane & 15;

    // Hoisted A fragments (Q: 12, GW: 2) — loaded once, reused for 8 tiles.
    v16bf qa[12];
#pragma unroll
    for (int j = 0; j < 12; ++j) qa[j] = load_a_bf(qtile, QKVG_, j * 32, lane);
    v16bf ga[2];
#pragma unroll
    for (int j = 0; j < 2; ++j) ga[j] = load_a_bf(gtile, GPP_, j * 32, lane);

    // ---- Phases A & B: 64 key tiles distributed over 8 waves ----
    for (int t = wave; t < 64; t += 8) {
        const unsigned short* krows = kbase + (size_t)t * 16 * QKVG_;
        __builtin_prefetch(krows + 8 * QKVG_, 0, 1);
        v8f c0 = zero8(), c1 = zero8();        // dual accumulators: break RAW chain
        v16bf b0 = load_b_kcontig(krows, QKVG_, 0, lane);
        v16bf b1 = load_b_kcontig(krows, QKVG_, 32, lane);
#pragma unroll
        for (int j = 0; j < 12; j += 2) {
            int jn = (j + 2 < 12) ? j + 2 : j;             // tail: cached repeat
            v16bf nb0 = load_b_kcontig(krows, QKVG_, jn * 32, lane);
            v16bf nb1 = load_b_kcontig(krows, QKVG_, (jn + 1) * 32, lane);
            c0 = WMMA_BF16(c0, qa[j], b0);
            c1 = WMMA_BF16(c1, qa[j + 1], b1);
            b0 = nb0; b1 = nb1;
        }
#pragma unroll
        for (int r = 0; r < 8; ++r)
            sS[(r + m0) * N_ + t * 16 + n] = (c0[r] + c1[r]) * scale;

        const unsigned short* grows = gall + (size_t)t * 16 * GPP_;
        v8f d0 = zero8(), d1 = zero8();
        {
            v16bf e0 = load_b_kcontig(grows, GPP_, 0, lane);
            v16bf e1 = load_b_kcontig(grows, GPP_, 32, lane);
            d0 = WMMA_BF16(d0, ga[0], e0);
            d1 = WMMA_BF16(d1, ga[1], e1);
        }
#pragma unroll
        for (int r = 0; r < 8; ++r)
            sG[(r + m0) * N_ + t * 16 + n] = d0[r] + d1[r];
    }
    __syncthreads();

    // ---- Phase C: per-row softmax + exact top-k mask + renormalize ----
    for (int rr = 0; rr < 2; ++rr) {
        int row = wave * 2 + rr;
        float* srow = sS + row * N_;
        float* grow = sG + row * N_;
        unsigned short* wrow = (unsigned short*)grow;   // reuse: wave owns this row

        float sv[32]; unsigned int gv[32];
#pragma unroll
        for (int i = 0; i < 32; ++i) {
            sv[i] = srow[lane + 32 * i];
            gv[i] = __float_as_uint(grow[lane + 32 * i]);  // nonneg -> monotonic bits
        }
        float mx = -1e30f;
#pragma unroll
        for (int i = 0; i < 32; ++i) mx = fmaxf(mx, sv[i]);
#pragma unroll
        for (int off = 1; off < 32; off <<= 1) mx = fmaxf(mx, __shfl_xor(mx, off, 32));
        float denom = 0.0f;
#pragma unroll
        for (int i = 0; i < 32; ++i) { sv[i] = __expf(sv[i] - mx); denom += sv[i]; }
#pragma unroll
        for (int off = 1; off < 32; off <<= 1) denom += __shfl_xor(denom, off, 32);

        // binary search for 204th-largest gsim value (uint bits, wave-uniform)
        unsigned int lo = 0u, hi = 0x7F800000u;
        while (lo < hi) {
            unsigned int mid = lo + ((hi - lo + 1u) >> 1);
            int cnt = 0;
#pragma unroll
            for (int i = 0; i < 32; ++i) cnt += (gv[i] >= mid) ? 1 : 0;
#pragma unroll
            for (int off = 1; off < 32; off <<= 1) cnt += __shfl_xor(cnt, off, 32);
            if (cnt >= TOPK_) lo = mid; else hi = mid - 1u;
        }
        unsigned int thr = lo;

        float wsum = 0.0f;
#pragma unroll
        for (int i = 0; i < 32; ++i) {
            float wv = (gv[i] >= thr) ? sv[i] : 0.0f;
            sv[i] = wv; wsum += wv;
        }
#pragma unroll
        for (int off = 1; off < 32; off <<= 1) wsum += __shfl_xor(wsum, off, 32);
        float inv = 1.0f / (wsum + 1e-8f * denom);
#pragma unroll
        for (int i = 0; i < 32; ++i)
            wrow[lane + 32 * i] = f2bf(sv[i] * inv);
    }
    __syncthreads();

    // ---- Phase D: attn_out = W(16x1024) @ V(1024x384) ----
    // Stage each 32-key V chunk transposed into LDS (sV[dim][key], 24KB,
    // overlaying dead sS); all 8 waves share it. A fragment reused across the
    // wave's 3 dim-tiles.
    unsigned short* sV = (unsigned short*)sS;
    const unsigned short* sW = (const unsigned short*)sG;   // bf16 rows, stride N_
    v8f acc[3];
#pragma unroll
    for (int i = 0; i < 3; ++i) acc[i] = zero8();

    for (int kk = 0; kk < N_; kk += 32) {
        __syncthreads();
        // 32 keys x 384 dims; each thread moves 6 groups of 8 contiguous dims.
        for (int idx = threadIdx.x; idx < 32 * 48; idx += 256) {
            int key = idx / 48;
            int g8  = (idx % 48) * 8;
            U4w d; d.u = *(const uint4*)(vbase + (size_t)(kk + key) * QKVG_ + g8);
#pragma unroll
            for (int j = 0; j < 8; ++j) sV[(g8 + j) * 32 + key] = d.s[j];
        }
        __syncthreads();
        v16bf a = load_a_bf(sW, N_, kk, lane);
        v16bf bm[3];
#pragma unroll
        for (int i = 0; i < 3; ++i)
            bm[i] = load_b_kcontig(sV + (size_t)((wave + 8 * i) * 16) * 32, 32, 0, lane);
#pragma unroll
        for (int i = 0; i < 3; ++i)
            acc[i] = WMMA_BF16(acc[i], a, bm[i]);
    }
    size_t orow = (size_t)rt * 16;
#pragma unroll
    for (int i = 0; i < 3; ++i) {
        int t = wave + 8 * i;
#pragma unroll
        for (int r = 0; r < 8; ++r)
            attn[(orow + r + m0) * DIM_ + t * 16 + n] = f2bf(acc[i][r]);
    }
}

// ---------------------------------------------------------------------------
// Kernel 4: out = attn @ W_proj + b_proj (f32 output)
// Block = 1 wave computing 8 column tiles; grid (3, 1024).
// ---------------------------------------------------------------------------
__global__ __launch_bounds__(32) void k_proj(const unsigned short* __restrict__ attn,
                                             const unsigned short* __restrict__ WT,
                                             const float* __restrict__ bp,
                                             float* __restrict__ out) {
    int lane = threadIdx.x;
    int cb = blockIdx.x, rt = blockIdx.y;
    const unsigned short* arows = attn + (size_t)rt * 16 * DIM_;
    const unsigned short* wbase = WT + (size_t)(cb * 128) * DIM_;

    v8f acc[8];
#pragma unroll
    for (int t = 0; t < 8; ++t) {
        float bv = bp[cb * 128 + t * 16 + (lane & 15)];
#pragma unroll
        for (int i = 0; i < 8; ++i) acc[t][i] = bv;
    }
    gemm_8tile_pipelined(arows, wbase, lane, acc);

    int m0 = (lane & 16) ? 8 : 0, n = lane & 15;
#pragma unroll
    for (int t = 0; t < 8; ++t)
#pragma unroll
        for (int r = 0; r < 8; ++r)
            out[((size_t)rt * 16 + r + m0) * DIM_ + cb * 128 + t * 16 + n] = acc[t][r];
}

// ---------------------------------------------------------------------------
extern "C" void kernel_launch(void* const* d_in, const int* in_sizes, int n_in,
                              void* d_out, int out_size, void* d_ws, size_t ws_size,
                              hipStream_t stream) {
    (void)in_sizes; (void)n_in; (void)out_size; (void)ws_size;
    const float* x     = (const float*)d_in[0];
    const float* Wqkvg = (const float*)d_in[1];
    const float* bqkvg = (const float*)d_in[2];
    const float* Wproj = (const float*)d_in[3];
    const float* bproj = (const float*)d_in[4];
    const float* Wgp   = (const float*)d_in[5];

    unsigned short* ws_xbf    = (unsigned short*)d_ws;                      // 12.6 MB
    unsigned short* ws_qkvg   = ws_xbf    + (size_t)B_ * N_ * DIM_;         // 50.3 MB
    unsigned short* ws_gw     = ws_qkvg   + (size_t)B_ * N_ * QKVG_;        //  2.1 MB
    unsigned short* ws_attn   = ws_gw     + (size_t)B_ * N_ * GPP_;         // 12.6 MB
    unsigned short* ws_WqkvgT = ws_attn   + (size_t)B_ * N_ * DIM_;         //  1.2 MB
    unsigned short* ws_WprojT = ws_WqkvgT + (size_t)QKVG_ * DIM_;           //  0.3 MB
    unsigned short* ws_WgpT   = ws_WprojT + (size_t)DIM_ * DIM_;            //  48 KB

    // one-time format prep (bf16 conversion + weight transposes)
    k_prep_x<<<(B_ * N_ * DIM_) / (256 * 4), 256, 0, stream>>>(x, ws_xbf);
    k_prep_wT<<<QKVG_, 128, 0, stream>>>(Wqkvg, ws_WqkvgT, DIM_, QKVG_);
    k_prep_wT<<<DIM_,  128, 0, stream>>>(Wproj, ws_WprojT, DIM_, DIM_);
    k_prep_wT<<<GPP_,  128, 0, stream>>>(Wgp,   ws_WgpT,   DIM_, 49);

    k_qkvg<<<dim3(QKVG_ / 128, (B_ * N_) / 16), 32, 0, stream>>>(ws_xbf, ws_WqkvgT, bqkvg, ws_qkvg);
    k_gw<<<(B_ * N_) / 16, 32, 0, stream>>>(ws_qkvg, ws_WgpT, ws_gw);

    size_t shmem = (size_t)16 * N_ * sizeof(float) * 2;                      // 128 KB
    k_attn<<<(B_ * N_) / 16, 256, shmem, stream>>>(ws_qkvg, ws_gw, ws_attn);

    k_proj<<<dim3(DIM_ / 128, (B_ * N_) / 16), 32, 0, stream>>>(ws_attn, ws_WprojT, bproj,
                                                                (float*)d_out);
}
